// RNN_78228534329817
// MI455X (gfx1250) — matrix operand order repaired and checked by reading, not compiled, not run on hardware
//
#include <hip/hip_runtime.h>
#include <stdint.h>

// Problem constants (from reference): T=512, B=32, I=512, H=1024
#define T_STEPS 512
#define BATCH   32
#define IN_DIM  512
#define HID     1024
#define G3H     3072   // 3*H

typedef __attribute__((ext_vector_type(16))) __bf16          v16bf;
typedef __attribute__((ext_vector_type(16))) unsigned short  v16u;
typedef __attribute__((ext_vector_type(8)))  unsigned short  v8u;
typedef __attribute__((ext_vector_type(8)))  float           v8f;

__device__ __forceinline__ unsigned short f2bf(float f) {
  unsigned u = __float_as_uint(f);
  u += 0x7FFFu + ((u >> 16) & 1u);           // round-to-nearest-even
  return (unsigned short)(u >> 16);
}
__device__ __forceinline__ float bf2f(unsigned short h) {
  return __uint_as_float(((unsigned)h) << 16);
}
__device__ __forceinline__ float sigmoid_(float x) { return 1.0f / (1.0f + __expf(-x)); }
__device__ __forceinline__ float tanh_(float x)    { return 2.0f / (1.0f + __expf(-2.0f * x)) - 1.0f; }

// CDNA5 16-bit A fragment (16x32, row-major src): lane holds row (lane&15);
// elems 0..7 = K[base..base+7], 8..15 = K[base+16..base+23], base = k0+((lane>=16)?8:0)
__device__ __forceinline__ v16bf load_afrag(const unsigned short* p /* row + k0 + lb8 */) {
  v8u lo = *(const v8u*)(p);
  v8u hi = *(const v8u*)(p + 16);
  v16u a;
#pragma unroll
  for (int i = 0; i < 8; ++i) { a[i] = lo[i]; a[i + 8] = hi[i]; }
  return __builtin_bit_cast(v16bf, a);
}
__device__ __forceinline__ v16bf load_bfrag(const unsigned short* p) {
  return __builtin_bit_cast(v16bf, *(const v16u*)p);
}
__device__ __forceinline__ v8f wmma_bf16(v16bf a, v16bf b, v8f c) {
  return __builtin_amdgcn_wmma_f32_16x16x32_bf16(false, a, false, b, (short)0, c, false, false);
}

// Async global->LDS copy (CDNA5, tracked by ASYNCcnt). lds_off is the byte
// offset within the workgroup LDS (dynamic LDS starts at 0: no static LDS here).
__device__ __forceinline__ void async_g2l_b128(unsigned lds_off, const void* gaddr) {
  asm volatile("global_load_async_to_lds_b128 %0, %1, off"
               :: "v"(lds_off), "v"(gaddr) : "memory");
}
__device__ __forceinline__ void wait_async0() {
  asm volatile("s_wait_asynccnt 0x0" ::: "memory");
}

// ---------------------------------------------------------------- init
__global__ void k_init(unsigned int* cnt, unsigned short* hbf) {
  int i = blockIdx.x * blockDim.x + threadIdx.x;
  if (i < 64) cnt[i] = 0u;
  for (int j = i; j < 2 * BATCH * HID; j += gridDim.x * blockDim.x) hbf[j] = 0;
}

// ------------------------------------------------------- f32 -> bf16 copy
__global__ void k_cvt_x(const float* __restrict__ x, unsigned short* __restrict__ xb, int n) {
  for (int i = blockIdx.x * blockDim.x + threadIdx.x; i < n; i += gridDim.x * blockDim.x)
    xb[i] = f2bf(x[i]);
}

// ------------------------- pack weight [Krows x 3072] into B-fragment-major bf16
// dest element index d = ((ntile*KI + kit)*32 + lane)*16 + e
// source: col = ntile*16 + (lane&15); k = kit*32 + ((lane&16)?8:0) + (e&7) + ((e&8)?16:0)
__global__ void k_pack_w(const float* __restrict__ W, unsigned short* __restrict__ P, int Krows) {
  const int KI = Krows >> 5;
  const int n = Krows * G3H;
  for (int d = blockIdx.x * blockDim.x + threadIdx.x; d < n; d += gridDim.x * blockDim.x) {
    int e     = d & 15;
    int lane  = (d >> 4) & 31;
    int kit   = (d >> 9) % KI;
    int ntile = d / (KI << 9);
    int col   = (ntile << 4) + (lane & 15);
    int k     = (kit << 5) + ((lane & 16) ? 8 : 0) + (e & 7) + ((e & 8) ? 16 : 0);
    P[d] = f2bf(W[(size_t)k * G3H + col]);
  }
}

// --------------------------------------------- GI = Xbf[16384x512] @ WihPack -> bf16
// Wave computes 16x64 (1 M-tile x 4 N-tiles). 1024 M-tiles * 48 N-quads = 49152 waves.
__global__ void __launch_bounds__(256) k_gi(const unsigned short* __restrict__ xb,
                                            const unsigned short* __restrict__ wp,
                                            unsigned short* __restrict__ gi) {
  const int gw   = blockIdx.x * 8 + (threadIdx.x >> 5);
  const int lane = threadIdx.x & 31;
  const int nl   = lane & 15;
  const int lb8  = (lane & 16) ? 8 : 0;
  const int mt = gw / 48;
  const int nq = gw - mt * 48;
  const int r0 = mt << 4;

  v8f acc[4];
#pragma unroll
  for (int q = 0; q < 4; ++q) acc[q] = (v8f){0, 0, 0, 0, 0, 0, 0, 0};

  const unsigned short* arow = xb + (size_t)(r0 + nl) * IN_DIM + lb8;
  for (int kit = 0; kit < 16; ++kit) {
    v16bf a = load_afrag(arow + (kit << 5));
#pragma unroll
    for (int q = 0; q < 4; ++q) {
      const unsigned short* bp =
          wp + ((((size_t)(nq * 4 + q) * 16 + kit) * 32 + lane) << 4);
      acc[q] = wmma_bf16(a, load_bfrag(bp), acc[q]);
    }
  }
  const int rbase = r0 + ((lane & 16) ? 8 : 0);
#pragma unroll
  for (int q = 0; q < 4; ++q) {
    int col = (nq << 6) + (q << 4) + nl;
#pragma unroll
    for (int v = 0; v < 8; ++v)
      gi[(size_t)(rbase + v) * G3H + col] = f2bf(acc[q][v]);
  }
}

// ----------------------------------------------------- persistent GRU recurrence
// 64 blocks x 2 waves. Block owns H-slice hs=blockIdx.x (16 cols); wave = M-tile.
// W_hh slice (3 gates x 32 kit x 1KB = 96KB) staged once into LDS via async
// global->LDS loads; all 512 steps read B-fragments from LDS. f32 state h[b][j]
// lives in registers (writer lane == reader lane); bf16 copy in global is the
// cross-wave WMMA A operand, double buffered. One grid barrier per step.
__global__ void __launch_bounds__(64, 1)
k_gru(const unsigned short* __restrict__ gi, const unsigned short* __restrict__ whp,
      const float* __restrict__ pad, const float* __restrict__ bih,
      const float* __restrict__ bhh, unsigned short* hbf,
      float* __restrict__ out, unsigned int* cnt) {
  extern __shared__ unsigned short smem[];   // 96 KB dynamic LDS
  const int tid  = threadIdx.x;
  const int lane = tid & 31;
  const int nl   = lane & 15;
  const int lb8  = (lane & 16) ? 8 : 0;
  const int hs   = blockIdx.x;               // 0..63 H-slice
  const int m0   = (tid >> 5) << 4;          // wave 0: rows 0..15, wave 1: 16..31
  const int n0   = hs << 4;
  const int rbase = m0 + lb8;                // C/D layout row base
  const int j     = n0 + nl;                 // this lane's H column
  const unsigned int nb = gridDim.x;

  // ---- stage this block's W_hh slice into LDS (async, 96KB, 6144 x 16B chunks)
  for (int c = tid; c < 6144; c += 64) {
    const int fragidx = c >> 1;              // (g,kit,lane2) fragment
    const int half    = c & 1;
    const int lane2   = fragidx & 31;
    const int kit2    = (fragidx >> 5) & 31;
    const int g2      = fragidx >> 10;       // 0..2
    const size_t goff = ((((size_t)(g2 * 64 + hs) * 32 + kit2) * 32 + lane2) << 4) + (half << 3);
    async_g2l_b128((unsigned)(c << 4), (const void*)(whp + goff));
  }
  wait_async0();
  __syncthreads();

  // ---- time-invariant per-lane values
  const float bihr = bih[j],            bhhr = bhh[j];
  const float bihz = bih[HID + j],      bhhz = bhh[HID + j];
  const float bihn = bih[2 * HID + j],  bhhn = bhh[2 * HID + j];
  float hold[8];
#pragma unroll
  for (int v = 0; v < 8; ++v) hold[v] = 0.0f;

  for (int t = 0; t < T_STEPS; ++t) {
    const unsigned short* hr = hbf + (size_t)(t & 1) * (BATCH * HID);        // read buf
    unsigned short*       hw = hbf + (size_t)((t + 1) & 1) * (BATCH * HID);  // write buf

    // issue the elementwise-phase loads early so they overlap the GEMM
    float p8[8], gir8[8], giz8[8], gin8[8];
#pragma unroll
    for (int v = 0; v < 8; ++v) {
      const int b = rbase + v;
      const size_t girow = ((size_t)t * BATCH + b) * G3H;
      gir8[v] = bf2f(gi[girow + j]);
      giz8[v] = bf2f(gi[girow + HID + j]);
      gin8[v] = bf2f(gi[girow + 2 * HID + j]);
      p8[v]   = pad[t * BATCH + b];
    }

    v8f acc[3];
#pragma unroll
    for (int g = 0; g < 3; ++g) acc[g] = (v8f){0, 0, 0, 0, 0, 0, 0, 0};

    const unsigned short* arow = hr + (size_t)(m0 + nl) * HID + lb8;
    v16bf a_cur = load_afrag(arow);
    for (int kit = 0; kit < 32; ++kit) {
      v16bf a_nxt = load_afrag(arow + ((((kit + 1) & 31)) << 5));  // prefetch next
#pragma unroll
      for (int g = 0; g < 3; ++g) {
        const unsigned short* bp = smem + (((g * 32 + kit) * 32 + lane) << 4);
        acc[g] = wmma_bf16(a_cur, load_bfrag(bp), acc[g]);
      }
      a_cur = a_nxt;
    }

    // gate math + state update, state held in registers
#pragma unroll
    for (int v = 0; v < 8; ++v) {
      const int b = rbase + v;
      const float r  = sigmoid_(gir8[v] + bihr + acc[0][v] + bhhr);
      const float z  = sigmoid_(giz8[v] + bihz + acc[1][v] + bhhz);
      const float nn = tanh_(gin8[v] + bihn + r * (acc[2][v] + bhhn));
      float hnew = (1.0f - z) * nn + z * hold[v];
      hnew = p8[v] * hold[v] + (1.0f - p8[v]) * hnew;   // hold state on padded steps
      hold[v] = hnew;
      hw[(size_t)b * HID + j] = f2bf(hnew);
      const size_t o = (size_t)t * (BATCH * HID) + (size_t)b * HID + j;
      out[o] = hnew;                                    // outputs
      out[(size_t)T_STEPS * BATCH * HID + o] = hnew;    // states
    }

    // device-wide barrier (monotonic counter, one arrival per block)
    __threadfence();
    __syncthreads();
    if (tid == 0) {
      const unsigned int target = (unsigned int)(t + 1) * nb;
      atomicAdd(cnt, 1u);
      while (__hip_atomic_load(cnt, __ATOMIC_ACQUIRE, __HIP_MEMORY_SCOPE_AGENT) < target) {
        __builtin_amdgcn_s_sleep(1);
      }
    }
    __syncthreads();
  }
}

// ---------------------------------------------------------------- launcher
extern "C" void kernel_launch(void* const* d_in, const int* in_sizes, int n_in,
                              void* d_out, int out_size, void* d_ws, size_t ws_size,
                              hipStream_t stream) {
  (void)in_sizes; (void)n_in; (void)out_size; (void)ws_size;
  const float* x   = (const float*)d_in[0];   // [T,B,I]
  const float* pad = (const float*)d_in[1];   // [T,B,1]
  const float* wih = (const float*)d_in[2];   // [I,3H]
  const float* whh = (const float*)d_in[3];   // [H,3H]
  const float* bih = (const float*)d_in[4];   // [3H]
  const float* bhh = (const float*)d_in[5];   // [3H]
  float* out = (float*)d_out;                 // outputs ++ states, each [T,B,H]

  char* ws = (char*)d_ws;
  size_t off = 0;
  unsigned int*   cnt  = (unsigned int*)(ws + off);   off += 256;
  unsigned short* hbf  = (unsigned short*)(ws + off); off += (size_t)2 * BATCH * HID * 2;
  unsigned short* xb   = (unsigned short*)(ws + off); off += (size_t)T_STEPS * BATCH * IN_DIM * 2;
  unsigned short* wihp = (unsigned short*)(ws + off); off += (size_t)IN_DIM * G3H * 2;
  unsigned short* whhp = (unsigned short*)(ws + off); off += (size_t)HID * G3H * 2;
  unsigned short* gi   = (unsigned short*)(ws + off); off += (size_t)T_STEPS * BATCH * G3H * 2;
  // total ws usage ~122 MB

  k_init  <<<64,   256, 0, stream>>>(cnt, hbf);
  k_cvt_x <<<2048, 256, 0, stream>>>(x, xb, T_STEPS * BATCH * IN_DIM);
  k_pack_w<<<2048, 256, 0, stream>>>(wih, wihp, IN_DIM);
  k_pack_w<<<2048, 256, 0, stream>>>(whh, whhp, HID);
  k_gi    <<<6144, 256, 0, stream>>>(xb, wihp, gi);
  k_gru   <<<64, 64, 98304, stream>>>(gi, whhp, pad, bih, bhh, hbf, out, cnt);
}
